// KernelSelfAttn_17351667876265
// MI455X (gfx1250) — matrix operand order repaired and checked by reading, not compiled, not run on hardware
//
#include <hip/hip_runtime.h>

// ---------------- problem constants (match reference) ----------------
#define B_    4
#define N_    8192
#define D_    512
#define H_    8
#define MTOT  (B_ * N_)          // 32768 rows
#define NCOL  2048               // virtual fused columns: [q 512 | k 512 | non_att 512 | v 512]
#define EPS_  1e-6f
#define SPLITK 32                // split-K chunks for the kv reduction (K = 8192)

#define ABUF  (64 * 40)          // A tile LDS: 64 x 32 (+8 pad)
#define BBUF  (32 * 72)          // B tile LDS: 32 x 64 (+8 pad)

typedef __attribute__((ext_vector_type(16))) __bf16 v16bf;
typedef __attribute__((ext_vector_type(8)))  float  v8f;

// ---------------- CDNA5 async global->LDS copy (ASYNCcnt path) ----------------
#if defined(__gfx1250__) && __has_builtin(__builtin_amdgcn_global_load_async_to_lds_b128)
#define ASYNC_LDS 1
#else
#define ASYNC_LDS 0
#endif

#if ASYNC_LDS
typedef int v4i_ __attribute__((vector_size(16)));            // gcc-style int4 (matches builtin proto)
typedef __attribute__((address_space(1))) v4i_* gv4i_ptr;     // global (AS1)
typedef __attribute__((address_space(3))) v4i_* lv4i_ptr;     // LDS (AS3)

__device__ __forceinline__ void cp_async16(void* l, const void* g) {
  // global_load_async_to_lds_b128: 16B/lane, L2 -> LDS, no VGPR staging
  __builtin_amdgcn_global_load_async_to_lds_b128((gv4i_ptr)g, (lv4i_ptr)l, 0, 0);
}
template <int N>
__device__ __forceinline__ void cp_async_wait() {
#if __has_builtin(__builtin_amdgcn_s_wait_asynccnt)
  __builtin_amdgcn_s_wait_asynccnt(N);
#else
  asm volatile("s_wait_asynccnt %0" :: "i"(N) : "memory");
#endif
}
#endif

// ---------------- helpers ----------------
__device__ __forceinline__ v8f v8f_zero() {
  v8f z;
#pragma unroll
  for (int i = 0; i < 8; ++i) z[i] = 0.0f;
  return z;
}

__device__ __forceinline__ float phi_elu1(float z) {
  return z > 0.0f ? z + 1.0f : __expf(z);   // elu(z)+1
}

__device__ __forceinline__ v8f wmma_bf16(v16bf a, v16bf b, v8f c) {
  return __builtin_amdgcn_wmma_f32_16x16x32_bf16(
      false, a, false, b, (short)0, c, false, false);
}

// A-fragment: 16x32 bf16, source row-major [M][K], ld = lda (ISA 7.12.2 layout)
__device__ __forceinline__ v16bf load_a_frag(const __bf16* src, int lda, int m0, int k0) {
  const int lane = threadIdx.x & 31;
  const int m  = m0 + (lane & 15);
  const int kb = k0 + ((lane >> 4) << 3);
  const __bf16* row = src + m * lda;
  v16bf a;
#pragma unroll
  for (int r = 0; r < 4; ++r) {
    a[2*r    ] = row[kb + 2*r];
    a[2*r + 1] = row[kb + 2*r + 1];
    a[2*r + 8] = row[kb + 16 + 2*r];
    a[2*r + 9] = row[kb + 16 + 2*r + 1];
  }
  return a;
}

// A-fragment from transposed storage: A[m][k] = srcT[k][m]
__device__ __forceinline__ v16bf load_at_frag(const __bf16* srcT, int ldt, int m0, int k0) {
  const int lane = threadIdx.x & 31;
  const int m  = m0 + (lane & 15);
  const int kb = k0 + ((lane >> 4) << 3);
  v16bf a;
#pragma unroll
  for (int r = 0; r < 4; ++r) {
    a[2*r    ] = srcT[(kb + 2*r     ) * ldt + m];
    a[2*r + 1] = srcT[(kb + 2*r + 1 ) * ldt + m];
    a[2*r + 8] = srcT[(kb + 16 + 2*r) * ldt + m];
    a[2*r + 9] = srcT[(kb + 17 + 2*r) * ldt + m];
  }
  return a;
}

// B-fragment: 32x16 bf16, source row-major [K][N]; lane -> N, element -> K (table order)
__device__ __forceinline__ v16bf load_b_frag(const __bf16* src, int ldb, int k0, int n0) {
  const int lane = threadIdx.x & 31;
  const int n  = n0 + (lane & 15);
  const int kb = k0 + ((lane >> 4) << 3);
  v16bf b;
#pragma unroll
  for (int e = 0; e < 8; ++e) {
    b[e]     = src[(kb + e) * ldb + n];
    b[e + 8] = src[(kb + 16 + e) * ldb + n];
  }
  return b;
}

// ---------------- utility kernels ----------------
__global__ void f32_to_bf16_kernel(const float* __restrict__ s, __bf16* __restrict__ d, int n) {
  for (int i = blockIdx.x * blockDim.x + threadIdx.x; i < n; i += gridDim.x * blockDim.x)
    d[i] = (__bf16)s[i];
}

__global__ void zero_f32_kernel(float* __restrict__ p, int n) {
  for (int i = blockIdx.x * blockDim.x + threadIdx.x; i < n; i += gridDim.x * blockDim.x)
    p[i] = 0.0f;
}

// ---------------- stage 1: fused h1/v GEMM + phi epilogue ----------------
__global__ void __launch_bounds__(128) gemm_qkv_kernel(
    const __bf16* __restrict__ xbf, const __bf16* __restrict__ W1bf,
    const __bf16* __restrict__ Wvbf, const float* __restrict__ b1,
    const float* __restrict__ bv, const unsigned char* __restrict__ mask,
    __bf16* __restrict__ qf, __bf16* __restrict__ kf, __bf16* __restrict__ vh,
    float* __restrict__ outF)
{
  __shared__ __align__(16) __bf16 As[2 * ABUF];   // double-buffered
  __shared__ __align__(16) __bf16 Bs[2 * BBUF];
  const int t = threadIdx.x;
  const int lane = t & 31, wid = t >> 5;
  const int wm = (wid >> 1) * 32, wn = (wid & 1) * 32;
  const int gm0 = blockIdx.x * 64;
  const int nv0 = blockIdx.y * 64;

  const __bf16* wsrc; int ldw, ncol;
  if (nv0 < 1536) { wsrc = W1bf; ldw = 1536; ncol = nv0; }
  else            { wsrc = Wvbf; ldw = 512;  ncol = nv0 - 1536; }

  const int rA = t >> 2, qA = (t & 3) * 8;       // A tile: 64 rows x 4 quads
  const int rB = t >> 3, qB = (t & 7) * 8;       // B tile: 32 rows x 8 quads
  const int aoff  = rA * 40 + qA,  aoff2 = (rA + 32) * 40 + qA;
  const int boff  = rB * 72 + qB,  boff2 = (rB + 16) * 72 + qB;
  const __bf16* gA0 = &xbf[(gm0 + rA) * 512 + qA];
  const __bf16* gA1 = &xbf[(gm0 + rA + 32) * 512 + qA];
  const __bf16* gB0 = &wsrc[rB * ldw + ncol + qB];
  const __bf16* gB1 = &wsrc[(rB + 16) * ldw + ncol + qB];

  v8f c[2][2];
#pragma unroll
  for (int i = 0; i < 2; ++i)
#pragma unroll
    for (int j = 0; j < 2; ++j) c[i][j] = v8f_zero();

#if ASYNC_LDS
  auto fill = [&](int buf, int k0) {
    cp_async16(&As[buf * ABUF + aoff ], gA0 + k0);
    cp_async16(&As[buf * ABUF + aoff2], gA1 + k0);
    cp_async16(&Bs[buf * BBUF + boff ], gB0 + k0 * ldw);
    cp_async16(&Bs[buf * BBUF + boff2], gB1 + k0 * ldw);
  };
  fill(0, 0);
  for (int it = 0; it < 16; ++it) {
    const int buf = it & 1;
    if (it < 15) { fill(buf ^ 1, (it + 1) * 32); cp_async_wait<4>(); }
    else         { cp_async_wait<0>(); }
    __syncthreads();
    const __bf16* Ab = &As[buf * ABUF];
    const __bf16* Bb = &Bs[buf * BBUF];
    v16bf a0  = load_a_frag(Ab, 40, wm, 0);
    v16bf a1  = load_a_frag(Ab, 40, wm + 16, 0);
    v16bf bf0 = load_b_frag(Bb, 72, 0, wn);
    v16bf bf1 = load_b_frag(Bb, 72, 0, wn + 16);
    c[0][0] = wmma_bf16(a0, bf0, c[0][0]);
    c[0][1] = wmma_bf16(a0, bf1, c[0][1]);
    c[1][0] = wmma_bf16(a1, bf0, c[1][0]);
    c[1][1] = wmma_bf16(a1, bf1, c[1][1]);
    __syncthreads();
  }
#else
  // fallback: single LDS buffer, register prefetch of the next tile during compute
  uint4 na0 = *(const uint4*)gA0;
  uint4 na1 = *(const uint4*)gA1;
  uint4 nb0 = *(const uint4*)gB0;
  uint4 nb1 = *(const uint4*)gB1;
  for (int it = 0; it < 16; ++it) {
    *(uint4*)&As[aoff] = na0; *(uint4*)&As[aoff2] = na1;
    *(uint4*)&Bs[boff] = nb0; *(uint4*)&Bs[boff2] = nb1;
    __syncthreads();
    if (it < 15) {
      int k0 = (it + 1) * 32;
      na0 = *(const uint4*)(gA0 + k0);
      na1 = *(const uint4*)(gA1 + k0);
      nb0 = *(const uint4*)(gB0 + k0 * ldw);
      nb1 = *(const uint4*)(gB1 + k0 * ldw);
    }
    v16bf a0  = load_a_frag(As, 40, wm, 0);
    v16bf a1  = load_a_frag(As, 40, wm + 16, 0);
    v16bf bf0 = load_b_frag(Bs, 72, 0, wn);
    v16bf bf1 = load_b_frag(Bs, 72, 0, wn + 16);
    c[0][0] = wmma_bf16(a0, bf0, c[0][0]);
    c[0][1] = wmma_bf16(a0, bf1, c[0][1]);
    c[1][0] = wmma_bf16(a1, bf0, c[1][0]);
    c[1][1] = wmma_bf16(a1, bf1, c[1][1]);
    __syncthreads();
  }
#endif

#pragma unroll
  for (int i = 0; i < 2; ++i)
#pragma unroll
    for (int j = 0; j < 2; ++j) {
#pragma unroll
      for (int r = 0; r < 8; ++r) {
        int row = gm0 + wm + i * 16 + r + ((lane >> 4) << 3);
        int gn  = nv0 + wn + j * 16 + (lane & 15);
        float val = c[i][j][r];
        float mk = mask[row] ? 1.0f : 0.0f;
        if (gn < 512) {
          val = phi_elu1(val + b1[gn]);
          qf[row * 512 + gn] = (__bf16)val;
        } else if (gn < 1024) {
          val = phi_elu1(val + b1[gn]) * mk;
          kf[row * 512 + (gn - 512)] = (__bf16)val;
        } else if (gn < 1536) {
          outF[row * 512 + (gn - 1024)] = val + b1[gn];
        } else {
          val = (val + bv[gn - 1536]) * mk;
          vh[row * 512 + (gn - 1536)] = (__bf16)val;
        }
      }
    }
}

// ---------------- stage 2: kv[b,h] = kf^T @ vh (64x64, K=8192) + ksum ----------------
__global__ void __launch_bounds__(128) kv_reduce_kernel(
    const __bf16* __restrict__ kf, const __bf16* __restrict__ vh,
    float* __restrict__ kv, float* __restrict__ ksum)
{
  __shared__ __align__(16) __bf16 Ks[2 * BBUF];   // 32 n x 64 f, double-buffered
  __shared__ __align__(16) __bf16 Vs[2 * BBUF];   // 32 n x 64 d
  const int t = threadIdx.x, lane = t & 31, wid = t >> 5;
  const int bh = blockIdx.x / SPLITK, chunk = blockIdx.x % SPLITK;
  const int b = bh / H_, h = bh % H_;
  const int base_n = chunk * (N_ / SPLITK);

  const int rT = t >> 3, qT = (t & 7) * 8;       // 32 rows x 8 quads per tile
  const int off  = rT * 72 + qT;
  const int off2 = (rT + 16) * 72 + qT;
  const __bf16* gK = &kf[(b * N_ + base_n + rT) * 512 + h * 64 + qT];
  const __bf16* gV = &vh[(b * N_ + base_n + rT) * 512 + h * 64 + qT];

  v8f c[4];
#pragma unroll
  for (int j = 0; j < 4; ++j) c[j] = v8f_zero();
  float ksl = 0.0f;

  const int NIT = (N_ / SPLITK) / 32;   // 8 iterations of 32 rows

#if ASYNC_LDS
  auto fillKV = [&](int buf, int nn) {
    const __bf16* k = gK + nn * 512;
    const __bf16* v = gV + nn * 512;
    cp_async16(&Ks[buf * BBUF + off ], k);
    cp_async16(&Ks[buf * BBUF + off2], k + 16 * 512);
    cp_async16(&Vs[buf * BBUF + off ], v);
    cp_async16(&Vs[buf * BBUF + off2], v + 16 * 512);
  };
  fillKV(0, 0);
  for (int it = 0; it < NIT; ++it) {
    const int buf = it & 1;
    if (it < NIT - 1) { fillKV(buf ^ 1, (it + 1) * 32); cp_async_wait<4>(); }
    else              { cp_async_wait<0>(); }
    __syncthreads();
    const __bf16* Kb = &Ks[buf * BBUF];
    const __bf16* Vb = &Vs[buf * BBUF];
    if (t < 64) {                       // ksum: column sums of the kf tile
      float s = 0.0f;
      for (int r = 0; r < 32; ++r) s += (float)Kb[r * 72 + t];
      ksl += s;
    }
    v16bf a = load_at_frag(Kb, 72, wid * 16, 0);   // A[f][n] = Ks[n][f]
#pragma unroll
    for (int j = 0; j < 4; ++j) {
      v16bf bfr = load_b_frag(Vb, 72, 0, j * 16);
      c[j] = wmma_bf16(a, bfr, c[j]);
    }
    __syncthreads();
  }
#else
  uint4 nk0 = *(const uint4*)gK;
  uint4 nk1 = *(const uint4*)(gK + 16 * 512);
  uint4 nv0 = *(const uint4*)gV;
  uint4 nv1 = *(const uint4*)(gV + 16 * 512);
  for (int it = 0; it < NIT; ++it) {
    *(uint4*)&Ks[off] = nk0; *(uint4*)&Ks[off2] = nk1;
    *(uint4*)&Vs[off] = nv0; *(uint4*)&Vs[off2] = nv1;
    __syncthreads();
    if (it < NIT - 1) {
      int nn = (it + 1) * 32;
      nk0 = *(const uint4*)(gK + nn * 512);
      nk1 = *(const uint4*)(gK + (nn + 16) * 512);
      nv0 = *(const uint4*)(gV + nn * 512);
      nv1 = *(const uint4*)(gV + (nn + 16) * 512);
    }
    if (t < 64) {
      float s = 0.0f;
      for (int r = 0; r < 32; ++r) s += (float)Ks[r * 72 + t];
      ksl += s;
    }
    v16bf a = load_at_frag(Ks, 72, wid * 16, 0);
#pragma unroll
    for (int j = 0; j < 4; ++j) {
      v16bf bfr = load_b_frag(Vs, 72, 0, j * 16);
      c[j] = wmma_bf16(a, bfr, c[j]);
    }
    __syncthreads();
  }
#endif

#pragma unroll
  for (int j = 0; j < 4; ++j) {
#pragma unroll
    for (int r = 0; r < 8; ++r) {
      int f = wid * 16 + r + ((lane >> 4) << 3);
      int d = j * 16 + (lane & 15);
      atomicAdd(&kv[(bh * 64 + f) * 64 + d], c[j][r]);
    }
  }
  if (t < 64) atomicAdd(&ksum[bh * 64 + t], ksl);
}

// ---------------- stage 3: num = qf@kv, den = qf.ksum, att = num/(den+eps) ----------------
__global__ void __launch_bounds__(128) attn_kernel(
    const __bf16* __restrict__ qf, const __bf16* __restrict__ kvbf,
    const float* __restrict__ ksum, __bf16* __restrict__ attb)
{
  __shared__ __align__(16) __bf16 Qs[64 * 72];
  __shared__ __align__(16) __bf16 KVs[64 * 72];
  __shared__ float dens[64];
  __shared__ float kss[64];
  const int t = threadIdx.x, lane = t & 31, wid = t >> 5;
  const int bh = blockIdx.x >> 7, rt = blockIdx.x & 127;
  const int b = bh / H_, h = bh % H_;
  const int n0 = rt * 64;

  const int rT = t >> 3, qT = (t & 7) * 8;       // 64 rows x 8 quads, 4 chunks
#if ASYNC_LDS
#pragma unroll
  for (int s = 0; s < 4; ++s) {
    int row = rT + s * 16;
    cp_async16(&Qs[row * 72 + qT],  &qf[(b * N_ + n0 + row) * 512 + h * 64 + qT]);
    cp_async16(&KVs[row * 72 + qT], &kvbf[(bh * 64 + row) * 64 + qT]);
  }
  cp_async_wait<0>();
#else
  uint4 rq[4], rk[4];
#pragma unroll
  for (int s = 0; s < 4; ++s) {
    int row = rT + s * 16;
    rq[s] = *(const uint4*)&qf[(b * N_ + n0 + row) * 512 + h * 64 + qT];
    rk[s] = *(const uint4*)&kvbf[(bh * 64 + row) * 64 + qT];
  }
#pragma unroll
  for (int s = 0; s < 4; ++s) {
    int row = rT + s * 16;
    *(uint4*)&Qs[row * 72 + qT]  = rq[s];
    *(uint4*)&KVs[row * 72 + qT] = rk[s];
  }
#endif
  if (t < 64) kss[t] = ksum[bh * 64 + t];
  __syncthreads();

  if (t < 64) {
    float s = 0.0f;
    for (int f = 0; f < 64; ++f) s += (float)Qs[t * 72 + f] * kss[f];
    dens[t] = s;
  }

  v8f c[4];
#pragma unroll
  for (int j = 0; j < 4; ++j) c[j] = v8f_zero();
#pragma unroll
  for (int ks = 0; ks < 64; ks += 32) {
    v16bf a = load_a_frag(Qs, 72, wid * 16, ks);
#pragma unroll
    for (int j = 0; j < 4; ++j) {
      v16bf bfr = load_b_frag(KVs, 72, ks, j * 16);
      c[j] = wmma_bf16(a, bfr, c[j]);
    }
  }
  __syncthreads();

#pragma unroll
  for (int j = 0; j < 4; ++j) {
#pragma unroll
    for (int r = 0; r < 8; ++r) {
      int m = wid * 16 + r + ((lane >> 4) << 3);
      int d = j * 16 + (lane & 15);
      float v = c[j][r] / (dens[m] + EPS_);
      attb[(b * N_ + n0 + m) * 512 + h * 64 + d] = (__bf16)v;
    }
  }
}

// ---------------- stage 4: out = (non_att + att@Wo + bo) * mask ----------------
__global__ void __launch_bounds__(128) gemm_out_kernel(
    const __bf16* __restrict__ attb, const __bf16* __restrict__ Wobf,
    const float* __restrict__ bo, const unsigned char* __restrict__ mask,
    float* __restrict__ outF)
{
  __shared__ __align__(16) __bf16 As[2 * ABUF];
  __shared__ __align__(16) __bf16 Bs[2 * BBUF];
  const int t = threadIdx.x;
  const int lane = t & 31, wid = t >> 5;
  const int wm = (wid >> 1) * 32, wn = (wid & 1) * 32;
  const int gm0 = blockIdx.x * 64;
  const int n0  = blockIdx.y * 64;

  const int rA = t >> 2, qA = (t & 3) * 8;
  const int rB = t >> 3, qB = (t & 7) * 8;
  const int aoff  = rA * 40 + qA,  aoff2 = (rA + 32) * 40 + qA;
  const int boff  = rB * 72 + qB,  boff2 = (rB + 16) * 72 + qB;
  const __bf16* gA0 = &attb[(gm0 + rA) * 512 + qA];
  const __bf16* gA1 = &attb[(gm0 + rA + 32) * 512 + qA];
  const __bf16* gB0 = &Wobf[rB * 512 + n0 + qB];
  const __bf16* gB1 = &Wobf[(rB + 16) * 512 + n0 + qB];

  v8f c[2][2];
#pragma unroll
  for (int i = 0; i < 2; ++i)
#pragma unroll
    for (int j = 0; j < 2; ++j) c[i][j] = v8f_zero();

#if ASYNC_LDS
  auto fill = [&](int buf, int k0) {
    cp_async16(&As[buf * ABUF + aoff ], gA0 + k0);
    cp_async16(&As[buf * ABUF + aoff2], gA1 + k0);
    cp_async16(&Bs[buf * BBUF + boff ], gB0 + k0 * 512);
    cp_async16(&Bs[buf * BBUF + boff2], gB1 + k0 * 512);
  };
  fill(0, 0);
  for (int it = 0; it < 16; ++it) {
    const int buf = it & 1;
    if (it < 15) { fill(buf ^ 1, (it + 1) * 32); cp_async_wait<4>(); }
    else         { cp_async_wait<0>(); }
    __syncthreads();
    const __bf16* Ab = &As[buf * ABUF];
    const __bf16* Bb = &Bs[buf * BBUF];
    v16bf a0  = load_a_frag(Ab, 40, wm, 0);
    v16bf a1  = load_a_frag(Ab, 40, wm + 16, 0);
    v16bf bf0 = load_b_frag(Bb, 72, 0, wn);
    v16bf bf1 = load_b_frag(Bb, 72, 0, wn + 16);
    c[0][0] = wmma_bf16(a0, bf0, c[0][0]);
    c[0][1] = wmma_bf16(a0, bf1, c[0][1]);
    c[1][0] = wmma_bf16(a1, bf0, c[1][0]);
    c[1][1] = wmma_bf16(a1, bf1, c[1][1]);
    __syncthreads();
  }
#else
  uint4 na0 = *(const uint4*)gA0;
  uint4 na1 = *(const uint4*)gA1;
  uint4 nb0 = *(const uint4*)gB0;
  uint4 nb1 = *(const uint4*)gB1;
  for (int it = 0; it < 16; ++it) {
    *(uint4*)&As[aoff] = na0; *(uint4*)&As[aoff2] = na1;
    *(uint4*)&Bs[boff] = nb0; *(uint4*)&Bs[boff2] = nb1;
    __syncthreads();
    if (it < 15) {
      int k0 = (it + 1) * 32;
      na0 = *(const uint4*)(gA0 + k0);
      na1 = *(const uint4*)(gA1 + k0);
      nb0 = *(const uint4*)(gB0 + k0 * 512);
      nb1 = *(const uint4*)(gB1 + k0 * 512);
    }
    v16bf a0  = load_a_frag(As, 40, wm, 0);
    v16bf a1  = load_a_frag(As, 40, wm + 16, 0);
    v16bf bf0 = load_b_frag(Bs, 72, 0, wn);
    v16bf bf1 = load_b_frag(Bs, 72, 0, wn + 16);
    c[0][0] = wmma_bf16(a0, bf0, c[0][0]);
    c[0][1] = wmma_bf16(a0, bf1, c[0][1]);
    c[1][0] = wmma_bf16(a1, bf0, c[1][0]);
    c[1][1] = wmma_bf16(a1, bf1, c[1][1]);
    __syncthreads();
  }
#endif

#pragma unroll
  for (int i = 0; i < 2; ++i)
#pragma unroll
    for (int j = 0; j < 2; ++j) {
#pragma unroll
      for (int r = 0; r < 8; ++r) {
        int row = gm0 + wm + i * 16 + r + ((lane >> 4) << 3);
        int gn  = n0 + wn + j * 16 + (lane & 15);
        float val = c[i][j][r] + bo[gn] + outF[row * 512 + gn];  // outF holds non_att
        outF[row * 512 + gn] = mask[row] ? val : 0.0f;
      }
    }
}

// ---------------- host launch ----------------
extern "C" void kernel_launch(void* const* d_in, const int* in_sizes, int n_in,
                              void* d_out, int out_size, void* d_ws, size_t ws_size,
                              hipStream_t stream) {
  (void)in_sizes; (void)n_in; (void)out_size; (void)ws_size;
  const float*         x    = (const float*)d_in[0];
  const unsigned char* mask = (const unsigned char*)d_in[1];   // jnp bool -> 1 byte
  const float*         W1   = (const float*)d_in[2];
  const float*         b1   = (const float*)d_in[3];
  const float*         Wv   = (const float*)d_in[4];
  const float*         bv   = (const float*)d_in[5];
  const float*         Wo   = (const float*)d_in[6];
  const float*         bo   = (const float*)d_in[7];
  float* outF = (float*)d_out;

  char* p = (char*)d_ws;
  auto alloc = [&](size_t bytes) -> char* {
    char* r = p;
    p += (bytes + 255) & ~(size_t)255;
    return r;
  };
  __bf16* xbf  = (__bf16*)alloc((size_t)MTOT * 512 * 2);
  __bf16* qf   = (__bf16*)alloc((size_t)MTOT * 512 * 2);
  __bf16* kff  = (__bf16*)alloc((size_t)MTOT * 512 * 2);
  __bf16* vhh  = (__bf16*)alloc((size_t)MTOT * 512 * 2);
  __bf16* attb = (__bf16*)alloc((size_t)MTOT * 512 * 2);
  __bf16* W1bf = (__bf16*)alloc((size_t)512 * 1536 * 2);
  __bf16* Wvbf = (__bf16*)alloc((size_t)512 * 512 * 2);
  __bf16* Wobf = (__bf16*)alloc((size_t)512 * 512 * 2);
  float*  kv   = (float*)alloc((size_t)B_ * H_ * 64 * 64 * 4);
  float*  ksum = (float*)alloc((size_t)B_ * H_ * 64 * 4);      // contiguous after kv
  __bf16* kvbf = (__bf16*)alloc((size_t)B_ * H_ * 64 * 64 * 2);

  f32_to_bf16_kernel<<<2048, 256, 0, stream>>>(x,  xbf,  MTOT * 512);
  f32_to_bf16_kernel<<<512,  256, 0, stream>>>(W1, W1bf, 512 * 1536);
  f32_to_bf16_kernel<<<256,  256, 0, stream>>>(Wv, Wvbf, 512 * 512);
  f32_to_bf16_kernel<<<256,  256, 0, stream>>>(Wo, Wobf, 512 * 512);

  zero_f32_kernel<<<64, 256, 0, stream>>>(kv, B_ * H_ * 64 * 64 + B_ * H_ * 64);

  gemm_qkv_kernel<<<dim3(MTOT / 64, NCOL / 64), dim3(128), 0, stream>>>(
      xbf, W1bf, Wvbf, b1, bv, mask, qf, kff, vhh, outF);

  kv_reduce_kernel<<<dim3(B_ * H_ * SPLITK), dim3(128), 0, stream>>>(kff, vhh, kv, ksum);

  f32_to_bf16_kernel<<<128, 256, 0, stream>>>(kv, kvbf, B_ * H_ * 64 * 64);

  attn_kernel<<<dim3(B_ * H_ * (N_ / 64)), dim3(128), 0, stream>>>(qf, kvbf, ksum, attb);

  gemm_out_kernel<<<dim3(MTOT / 64, 512 / 64), dim3(128), 0, stream>>>(
      attb, Wobf, bo, mask, outF);
}
